// LSTM_Vanilla_9320079033013
// MI455X (gfx1250) — matrix operand order repaired and checked by reading, not compiled, not run on hardware
//
#include <hip/hip_runtime.h>
#include <hip/hip_bf16.h>

typedef __attribute__((ext_vector_type(16))) __bf16 v16bf;
typedef __attribute__((ext_vector_type(8)))  __bf16 v8bf;
typedef __attribute__((ext_vector_type(8)))  float  v8f;

#define B_DIM 32768
#define H_DIM 512
#define K_DIM 1024   // I + H

struct U32x8 { uint4 a, b; };

static __device__ __forceinline__ v16bf make_frag(uint4 lo, uint4 hi) {
    U32x8 u;
    u.a = lo; u.b = hi;
    return __builtin_bit_cast(v16bf, u);
}

static __device__ __forceinline__ v8bf cvt8(float4 f0, float4 f1) {
    v8bf v;
    v[0] = (__bf16)f0.x; v[1] = (__bf16)f0.y;
    v[2] = (__bf16)f0.z; v[3] = (__bf16)f0.w;
    v[4] = (__bf16)f1.x; v[5] = (__bf16)f1.y;
    v[6] = (__bf16)f1.z; v[7] = (__bf16)f1.w;
    return v;
}

static __device__ __forceinline__ float fsig(float x) {
    float e = __builtin_amdgcn_exp2f(x * -1.4426950408889634f);
    return __builtin_amdgcn_rcpf(1.0f + e);
}

static __device__ __forceinline__ float ftanh(float x) {
    float e = __builtin_amdgcn_exp2f(x * -2.8853900817779268f);
    return (1.0f - e) * __builtin_amdgcn_rcpf(1.0f + e);
}

// ---------------------------------------------------------------------------
// Kernel 1: pack Wx[4,512,512] + Wh[4,512,512] (fp32) into bf16
// Wp[4][512][1024] (K-major), and pre-sum biases into bsum[4][512].
// ---------------------------------------------------------------------------
__global__ void pack_weights(const float* __restrict__ Wx,
                             const float* __restrict__ Wh,
                             const float* __restrict__ bx,
                             const float* __restrict__ bh,
                             __bf16* __restrict__ Wp,
                             float* __restrict__ bsum) {
    const long t    = (long)blockIdx.x * blockDim.x + threadIdx.x;
    const long base = t * 8;                       // 8 contiguous k per thread
    if (base < (long)4 * H_DIM * K_DIM) {
        const int  k  = (int)(base & (K_DIM - 1));
        const long gn = base >> 10;                // g*512 + n
        const float* src = (k < 512) ? (Wx + gn * 512 + k)
                                     : (Wh + gn * 512 + (k - 512));
        float4 f0 = *(const float4*)(src);
        float4 f1 = *(const float4*)(src + 4);
        *(v8bf*)(Wp + base) = cvt8(f0, f1);
    }
    if (t < 4 * H_DIM) bsum[t] = bx[t] + bh[t];
}

// ---------------------------------------------------------------------------
// Kernel 2: fused LSTM cell, double-buffered LDS A-staging.
// Block = 256 threads (8 waves). Output tile 64 (batch) x 64 (hidden).
// Wave (mwave 0..1, nwave 0..3): 32 rows x 16 cols x 4 gates.
// 32 K-steps of v_wmma_f32_16x16x32_bf16, one s_barrier per step.
// ---------------------------------------------------------------------------
__global__ __launch_bounds__(256)
void lstm_wmma(const float* __restrict__ x,
               const float* __restrict__ h_prev,
               const float* __restrict__ c_prev,
               const __bf16* __restrict__ Wp,
               const float*  __restrict__ bsum,
               float* __restrict__ out) {
    __shared__ __align__(16) __bf16 As[2][64][40];  // ping-pong, pad 32->40

    const int tid   = threadIdx.x;
    const int lane  = tid & 31;
    const int wid   = tid >> 5;
    const int r     = lane & 15;
    const int kh    = lane >> 4;       // which K-half this lane carries
    const int mwave = wid & 1;         // 0..1
    const int nwave = wid >> 1;        // 0..3

    const long m_blk = (long)blockIdx.y * 64;
    const int  n0    = blockIdx.x * 64 + nwave * 16;
    const int  ncol  = n0 + r;         // this lane's N column

    // --- init accumulators with bias ---
    v8f acc[2][4];
#pragma unroll
    for (int g = 0; g < 4; ++g) {
        float bv = bsum[g * H_DIM + ncol];
#pragma unroll
        for (int mt = 0; mt < 2; ++mt)
#pragma unroll
            for (int e = 0; e < 8; ++e) acc[mt][g][e] = bv;
    }

    // cooperative A staging: each thread moves 8 fp32 -> 8 bf16 per step
    const int  lr   = tid >> 2;                 // 0..63
    const int  lk   = (tid & 3) * 8;            // 0,8,16,24
    const long aoff = (m_blk + lr) * 512 + lk;  // row-base into x / h_prev

    // this lane's weight base: row ncol, K-major bf16
    const __bf16* wbase = Wp + (long)ncol * K_DIM + kh * 8;

    // --- prologue: stage tile 0 (kb=0, from x) into buffer 0 ---
    {
        float4 f0 = *(const float4*)(x + aoff);
        float4 f1 = *(const float4*)(x + aoff + 4);
        *(v8bf*)&As[0][lr][lk] = cvt8(f0, f1);
    }
    __syncthreads();

    for (int it = 0; it < 32; ++it) {
        const int  buf      = it & 1;
        const bool has_next = (it != 31);

        // ---- issue next tile's global load early (uniform ptr select) ----
        float4 f0{}, f1{};
        if (has_next) {
            const int kb_n = (it + 1) * 32;
            const float* ap = ((kb_n < 512) ? (x + kb_n)
                                            : (h_prev + (kb_n - 512))) + aoff;
            f0 = *(const float4*)(ap);
            f1 = *(const float4*)(ap + 4);
        }

        // ---- A fragments from LDS[buf] (A layout: interleaved 8s) ----
        v16bf afrag[2];
#pragma unroll
        for (int mt = 0; mt < 2; ++mt) {
            const __bf16* a0 = &As[buf][mwave * 32 + mt * 16 + r][kh * 8];
            uint4 lo = *(const uint4*)(a0);        // K = kh*8 + 0..7
            uint4 hi = *(const uint4*)(a0 + 16);   // K = 16 + kh*8 + 0..7
            afrag[mt] = make_frag(lo, hi);
        }

        // ---- B fragments from global (bf16 weights, L2-resident) ----
        v16bf bfrag[4];
#pragma unroll
        for (int g = 0; g < 4; ++g) {
            const __bf16* bp = wbase + (long)g * H_DIM * K_DIM + it * 32;
            uint4 lo = *(const uint4*)(bp);        // K = kh*8 + 0..7
            uint4 hi = *(const uint4*)(bp + 16);   // K = 16 + kh*8 + 0..7
            bfrag[g] = make_frag(lo, hi);
        }

        // ---- stage next tile into the other buffer ----
        if (has_next)
            *(v8bf*)&As[buf ^ 1][lr][lk] = cvt8(f0, f1);

        // ---- 8 WMMAs ----
#pragma unroll
        for (int mt = 0; mt < 2; ++mt)
#pragma unroll
            for (int g = 0; g < 4; ++g)
                acc[mt][g] = __builtin_amdgcn_wmma_f32_16x16x32_bf16(
                    false, afrag[mt], false, bfrag[g],
                    (short)0, acc[mt][g], false, false);

        __syncthreads();
    }

    // ---- elementwise LSTM epilogue ----
    const long mbase  = m_blk + mwave * 32 + kh * 8;  // C layout: M = e + 8*kh
    const long hc_off = (long)B_DIM * H_DIM;
#pragma unroll
    for (int mt = 0; mt < 2; ++mt) {
#pragma unroll
        for (int e = 0; e < 8; ++e) {
            long  row = mbase + mt * 16 + e;
            float ig  = fsig(acc[mt][0][e]);
            float fg  = fsig(acc[mt][1][e]);
            float gg  = ftanh(acc[mt][2][e]);
            float og  = fsig(acc[mt][3][e]);
            float cp  = c_prev[row * H_DIM + ncol];
            float cn  = ig * gg + fg * cp;
            float hn  = og * ftanh(cn);
            out[row * H_DIM + ncol]          = hn;   // h_next
            out[hc_off + row * H_DIM + ncol] = cn;   // c_next
        }
    }
}

extern "C" void kernel_launch(void* const* d_in, const int* in_sizes, int n_in,
                              void* d_out, int out_size, void* d_ws, size_t ws_size,
                              hipStream_t stream) {
    const float* x      = (const float*)d_in[0];
    const float* h_prev = (const float*)d_in[1];
    const float* c_prev = (const float*)d_in[2];
    const float* Wx     = (const float*)d_in[3];
    const float* bx     = (const float*)d_in[4];
    const float* Wh     = (const float*)d_in[5];
    const float* bh     = (const float*)d_in[6];

    __bf16* Wp   = (__bf16*)d_ws;                                 // 4 MB bf16
    float*  bsum = (float*)((char*)d_ws + (size_t)4 * H_DIM * K_DIM * 2);

    pack_weights<<<1024, 256, 0, stream>>>(Wx, Wh, bx, bh, Wp, bsum);

    dim3 grid(H_DIM / 64, B_DIM / 64);   // (8, 512)
    lstm_wmma<<<grid, 256, 0, stream>>>(x, h_prev, c_prev, Wp, bsum,
                                        (float*)d_out);
}
